// RiemmanianAttention_76132590289080
// MI455X (gfx1250) — matrix operand order repaired and checked by reading, not compiled, not run on hardware
//
#include <hip/hip_runtime.h>
#include <hip/hip_bf16.h>
#include <stdint.h>

typedef __bf16 bf16;
typedef __attribute__((ext_vector_type(16))) __bf16 v16bf;
typedef __attribute__((ext_vector_type(8)))  float  v8f;

#define WMMA_BF16(a, b, c) \
  __builtin_amdgcn_wmma_f32_16x16x32_bf16(false, (a), false, (b), (short)0, (c), false, false)

// Problem constants
#define Bsz 4
#define Nn  1024
#define Cc  768
#define Hh  12
#define Dd  64

union FragU { v16bf v; uint32_t u[8]; };

// A fragment: 16x32 bf16 tile, row-major, K contiguous (ISA 7.12.2 16-bit A layout)
__device__ inline v16bf frag_a(const bf16* A, int lda, int lane) {
  const int half = (lane >> 4) & 1;
  const bf16* row = A + (lane & 15) * lda;
  FragU f;
#pragma unroll
  for (int i = 0; i < 8; ++i) {
    int k = (i < 4) ? (half * 8 + 2 * i) : (16 + half * 8 + 2 * (i - 4));
    f.u[i] = *reinterpret_cast<const uint32_t*>(row + k);
  }
  return f.v;
}

// B fragment: logical B[k][n] = Bt[n][k], Bt row-major NxK, K contiguous
__device__ inline v16bf frag_bT(const bf16* Bt, int ldb, int lane) {
  const int half = (lane >> 4) & 1;
  const bf16* row = Bt + (lane & 15) * ldb;
  FragU f;
#pragma unroll
  for (int i = 0; i < 8; ++i)
    f.u[i] = *reinterpret_cast<const uint32_t*>(row + ((i < 4) ? (half * 8 + 2 * i) : (16 + half * 8 + 2 * (i - 4))));
  return f.v;
}

// ---------------- K0: fp32 -> bf16 conversion ----------------
__global__ void cvt_f32_bf16(const float* __restrict__ src, bf16* __restrict__ dst, int n) {
  int i = blockIdx.x * blockDim.x + threadIdx.x;
  int stride = gridDim.x * blockDim.x;
  for (; i < n; i += stride) dst[i] = (bf16)src[i];
}

// ---------------- K1: QKV GEMM + bias + center(q,k) + transpose(v) ----------------
// grid (256, 12): 16-row block of the 4096 token rows x head
__global__ __launch_bounds__(128) void qkv_kernel(
    const bf16* __restrict__ xb,     // [4096][768]
    const bf16* __restrict__ wq,     // [2304][768]
    const float* __restrict__ qkvb,  // [2304]
    bf16* __restrict__ qc,           // [B*H][N][D] centered
    bf16* __restrict__ kc,           // [B*H][N][D] centered
    bf16* __restrict__ vT)           // [B*H][D][N]
{
  const int nb = blockIdx.x;
  const int h  = blockIdx.y;
  const int wave = threadIdx.x >> 5;
  const int lane = threadIdx.x & 31;
  const int half = lane >> 4;
  const int row0 = nb * 16;

  __shared__ float qs[16 * 64], ks[16 * 64], vs[16 * 64];
  __shared__ float meanq[16], meank[16];

  v8f accq = {0.f,0.f,0.f,0.f,0.f,0.f,0.f,0.f};
  v8f acck = accq, accv = accq;

  const bf16* Abase = xb + (size_t)row0 * Cc;
  const int colq = h * 64 + wave * 16;
  const bf16* Bq = wq + (size_t)(colq) * Cc;
  const bf16* Bk = wq + (size_t)(Cc + colq) * Cc;
  const bf16* Bv = wq + (size_t)(2 * Cc + colq) * Cc;

  for (int k0 = 0; k0 < Cc; k0 += 32) {
    v16bf a  = frag_a (Abase + k0, Cc, lane);
    v16bf bq = frag_bT(Bq + k0, Cc, lane);
    v16bf bk = frag_bT(Bk + k0, Cc, lane);
    v16bf bv = frag_bT(Bv + k0, Cc, lane);
    accq = WMMA_BF16(a, bq, accq);
    acck = WMMA_BF16(a, bk, acck);
    accv = WMMA_BF16(a, bv, accv);
  }

  const int cl = wave * 16 + (lane & 15);   // 0..63 within head
#pragma unroll
  for (int r = 0; r < 8; ++r) {
    int rr = r + 8 * half;
    qs[rr * 64 + cl] = accq[r] + qkvb[h * 64 + cl];
    ks[rr * 64 + cl] = acck[r] + qkvb[Cc + h * 64 + cl];
    vs[rr * 64 + cl] = accv[r] + qkvb[2 * Cc + h * 64 + cl];
  }
  __syncthreads();
  if (threadIdx.x < 16) {
    float sq = 0.f, sk = 0.f;
    for (int d = 0; d < 64; ++d) { sq += qs[threadIdx.x * 64 + d]; sk += ks[threadIdx.x * 64 + d]; }
    meanq[threadIdx.x] = sq * (1.0f / 64.0f);
    meank[threadIdx.x] = sk * (1.0f / 64.0f);
  }
  __syncthreads();
  for (int idx = threadIdx.x; idx < 16 * 64; idx += 128) {
    int r = idx >> 6, d = idx & 63;
    int ng = row0 + r;
    int b = ng >> 10, n = ng & 1023;
    size_t bh = (size_t)b * Hh + h;
    qc[(bh * Nn + n) * Dd + d] = (bf16)(qs[idx] - meanq[r]);
    kc[(bh * Nn + n) * Dd + d] = (bf16)(ks[idx] - meank[r]);
    vT[(bh * Dd + d) * Nn + n] = (bf16)vs[idx];
  }
}

// ---------------- K2: fused |cov(q)-cov(k)|/63 -> softmax -> P @ V ----------------
// grid (64, 48): 16-row block x (b,h)
__global__ __launch_bounds__(128) void attn_kernel(
    const bf16* __restrict__ qc, const bf16* __restrict__ kc,
    const bf16* __restrict__ vT,
    float* __restrict__ att,      // [B*H][N][N]
    bf16* __restrict__ outh)      // [B][N][C]
{
  const int rb = blockIdx.x;
  const int bh = blockIdx.y;
  const int b = bh / Hh, h = bh % Hh;
  const int wave = threadIdx.x >> 5, lane = threadIdx.x & 31;
  const int half = lane >> 4;
  const int row0 = rb * 16;

  __shared__ bf16  sp[16 * Nn];      // 32 KB score/prob strip
  __shared__ float wred[4][16];
  __shared__ float red8[16][8];
  __shared__ float rowmax[16], rowinv[16];

  const bf16* Qb = qc + (size_t)bh * Nn * Dd;
  const bf16* Kb = kc + (size_t)bh * Nn * Dd;

  // A fragments for this row block (reused for all 64 column tiles)
  v16bf aq0 = frag_a(Qb + (size_t)row0 * Dd +  0, Dd, lane);
  v16bf aq1 = frag_a(Qb + (size_t)row0 * Dd + 32, Dd, lane);
  v16bf ak0 = frag_a(Kb + (size_t)row0 * Dd +  0, Dd, lane);
  v16bf ak1 = frag_a(Kb + (size_t)row0 * Dd + 32, Dd, lane);

  float vmax[8];
#pragma unroll
  for (int r = 0; r < 8; ++r) vmax[r] = -3.0e38f;

  for (int jt = wave; jt < Nn / 16; jt += 4) {
    const int j0 = jt * 16;
    v16bf bq0 = frag_bT(Qb + (size_t)j0 * Dd +  0, Dd, lane);
    v16bf bq1 = frag_bT(Qb + (size_t)j0 * Dd + 32, Dd, lane);
    v16bf bk0 = frag_bT(Kb + (size_t)j0 * Dd +  0, Dd, lane);
    v16bf bk1 = frag_bT(Kb + (size_t)j0 * Dd + 32, Dd, lane);
    v8f cq = {0.f,0.f,0.f,0.f,0.f,0.f,0.f,0.f};
    v8f ck = cq;
    cq = WMMA_BF16(aq0, bq0, cq);
    cq = WMMA_BF16(aq1, bq1, cq);
    ck = WMMA_BF16(ak0, bk0, ck);
    ck = WMMA_BF16(ak1, bk1, ck);
    const int col = j0 + (lane & 15);
#pragma unroll
    for (int r = 0; r < 8; ++r) {
      float s = fabsf(cq[r] - ck[r]) * (1.0f / 63.0f);
      vmax[r] = fmaxf(vmax[r], s);
      sp[(r + 8 * half) * Nn + col] = (bf16)s;
    }
  }
  // row max: reduce across the 16 lanes that share each row, then across waves
#pragma unroll
  for (int r = 0; r < 8; ++r) {
    float v = vmax[r];
    v = fmaxf(v, __shfl_xor(v, 1, 32));
    v = fmaxf(v, __shfl_xor(v, 2, 32));
    v = fmaxf(v, __shfl_xor(v, 4, 32));
    v = fmaxf(v, __shfl_xor(v, 8, 32));
    if ((lane & 15) == 0) wred[wave][r + 8 * half] = v;
  }
  __syncthreads();
  if (threadIdx.x < 16)
    rowmax[threadIdx.x] = fmaxf(fmaxf(wred[0][threadIdx.x], wred[1][threadIdx.x]),
                                fmaxf(wred[2][threadIdx.x], wred[3][threadIdx.x]));
  __syncthreads();
  // exp + partial row sums (exp stored back in bf16)
  {
    const int r = threadIdx.x >> 3, sub = threadIdx.x & 7;
    const float m = rowmax[r];
    float s = 0.f;
    for (int c = sub; c < Nn; c += 8) {
      float e = __expf((float)sp[r * Nn + c] - m);
      sp[r * Nn + c] = (bf16)e;
      s += e;
    }
    red8[r][sub] = s;
  }
  __syncthreads();
  if (threadIdx.x < 16) {
    float s = 0.f;
#pragma unroll
    for (int i = 0; i < 8; ++i) s += red8[threadIdx.x][i];
    rowinv[threadIdx.x] = 1.0f / s;
  }
  __syncthreads();
  // normalize: write att (f32, coalesced) once, keep bf16 probs in LDS
  {
    float* attRow = att + ((size_t)bh * Nn + row0) * Nn;
    for (int idx = threadIdx.x; idx < 16 * Nn; idx += 128) {
      int r = idx >> 10;
      float p = (float)sp[idx] * rowinv[r];
      attRow[(size_t)r * Nn + (idx & 1023)] = p;
      sp[idx] = (bf16)p;
    }
  }
  __syncthreads();
  // out = P @ V : wave w owns d-slice [w*16, w*16+16)
  const bf16* Vb = vT + ((size_t)bh * Dd + wave * 16) * Nn;
  v8f acc = {0.f,0.f,0.f,0.f,0.f,0.f,0.f,0.f};
  for (int k0 = 0; k0 < Nn; k0 += 32) {
    v16bf a  = frag_a (sp + k0, Nn, lane);   // from LDS
    v16bf bv = frag_bT(Vb + k0, Nn, lane);
    acc = WMMA_BF16(a, bv, acc);
  }
  const int dcol = h * 64 + wave * 16 + (lane & 15);
#pragma unroll
  for (int r = 0; r < 8; ++r) {
    int n = row0 + r + 8 * half;
    outh[((size_t)b * Nn + n) * Cc + dcol] = (bf16)acc[r];
  }
}

// ---------------- K3: proj GEMM + bias ----------------
// grid (256, 12)
__global__ __launch_bounds__(128) void proj_kernel(
    const bf16* __restrict__ outh,  // [4096][768]
    const bf16* __restrict__ pw,    // [768][768]
    const float* __restrict__ pb,   // [768]
    float* __restrict__ out)        // [4096][768]
{
  const int nb = blockIdx.x;
  const int cb = blockIdx.y;
  const int wave = threadIdx.x >> 5, lane = threadIdx.x & 31, half = lane >> 4;
  const int row0 = nb * 16;
  const int col0 = cb * 64 + wave * 16;
  const bf16* A  = outh + (size_t)row0 * Cc;
  const bf16* Bt = pw   + (size_t)col0 * Cc;
  v8f acc = {0.f,0.f,0.f,0.f,0.f,0.f,0.f,0.f};
  for (int k0 = 0; k0 < Cc; k0 += 32) {
    v16bf a  = frag_a (A + k0, Cc, lane);
    v16bf bw = frag_bT(Bt + k0, Cc, lane);
    acc = WMMA_BF16(a, bw, acc);
  }
  const int col = col0 + (lane & 15);
  const float bias = pb[col];
#pragma unroll
  for (int r = 0; r < 8; ++r)
    out[(size_t)(row0 + r + 8 * half) * Cc + col] = acc[r] + bias;
}

extern "C" void kernel_launch(void* const* d_in, const int* in_sizes, int n_in,
                              void* d_out, int out_size, void* d_ws, size_t ws_size,
                              hipStream_t stream) {
  const float* x      = (const float*)d_in[0];
  const float* qkv_w  = (const float*)d_in[1];
  const float* qkv_b  = (const float*)d_in[2];
  const float* proj_w = (const float*)d_in[3];
  const float* proj_b = (const float*)d_in[4];

  float* out = (float*)d_out;                       // [B,N,C]
  float* att = out + (size_t)Bsz * Nn * Cc;         // [B,H,N,N]

  char* ws = (char*)d_ws;
  bf16* xb   = (bf16*)ws; ws += (size_t)Bsz * Nn * Cc * 2;        // 6 MB
  bf16* wq   = (bf16*)ws; ws += (size_t)3 * Cc * Cc * 2;          // 3.4 MB
  bf16* pw   = (bf16*)ws; ws += (size_t)Cc * Cc * 2;              // 1.1 MB
  bf16* qc   = (bf16*)ws; ws += (size_t)Bsz * Hh * Nn * Dd * 2;   // 6 MB
  bf16* kc   = (bf16*)ws; ws += (size_t)Bsz * Hh * Nn * Dd * 2;   // 6 MB
  bf16* vT   = (bf16*)ws; ws += (size_t)Bsz * Hh * Dd * Nn * 2;   // 6 MB
  bf16* outh = (bf16*)ws; ws += (size_t)Bsz * Nn * Cc * 2;        // 6 MB

  (void)in_sizes; (void)n_in; (void)out_size; (void)ws_size;

  int nx = Bsz * Nn * Cc;
  int nw = 3 * Cc * Cc;
  int np = Cc * Cc;
  cvt_f32_bf16<<<dim3((nx + 255) / 256), dim3(256), 0, stream>>>(x, xb, nx);
  cvt_f32_bf16<<<dim3((nw + 255) / 256), dim3(256), 0, stream>>>(qkv_w, wq, nw);
  cvt_f32_bf16<<<dim3((np + 255) / 256), dim3(256), 0, stream>>>(proj_w, pw, np);

  qkv_kernel <<<dim3(256, 12), dim3(128), 0, stream>>>(xb, wq, qkv_b, qc, kc, vT);
  attn_kernel<<<dim3(64, 48),  dim3(128), 0, stream>>>(qc, kc, vT, att, outh);
  proj_kernel<<<dim3(256, 12), dim3(128), 0, stream>>>(outh, pw, proj_b, out);
}
